// BitNetLinear_39668317946068
// MI455X (gfx1250) — matrix operand order repaired and checked by reading, not compiled, not run on hardware
//
#include <hip/hip_runtime.h>

typedef _Float16 f16_t;
typedef __attribute__((ext_vector_type(16))) _Float16 v16h;
typedef __attribute__((ext_vector_type(8)))  _Float16 v8h;
typedef __attribute__((ext_vector_type(4)))  _Float16 v4h;
typedef __attribute__((ext_vector_type(8)))  float    v8f;
typedef __attribute__((ext_vector_type(4)))  float    v4f;

// ---------------------------------------------------------------------------
// CDNA5 async global->LDS copy (ASYNCcnt-tracked), per-lane 16B transfer.
// VDST = per-lane LDS byte address, VADDR = 64-bit global address (GV mode).
// ---------------------------------------------------------------------------
__device__ __forceinline__ unsigned lds_off(const void* p) {
    // generic pointer to LDS: low 32 bits are the wave-relative LDS byte offset
    return (unsigned)(unsigned long long)p;
}
__device__ __forceinline__ void async_load_b128(unsigned lds_addr, const void* gptr) {
    asm volatile("global_load_async_to_lds_b128 %0, %1, off"
                 :: "v"(lds_addr), "v"((unsigned long long)gptr)
                 : "memory");
}
__device__ __forceinline__ void wait_async0() {
    asm volatile("s_wait_asynccnt 0x0" ::: "memory");
}

// ---------------------------------------------------------------------------
// Phase 1: one-time conversions into workspace
// ---------------------------------------------------------------------------
__global__ __launch_bounds__(256)
void cvt_x_f16(const float* __restrict__ X, f16_t* __restrict__ Xh, long n4)
{
    long i = (long)blockIdx.x * 256 + threadIdx.x;
    if (i < n4) {
        v4f v = *(const v4f*)&X[i * 4];
        v4h h;
        h.x = (f16_t)v.x; h.y = (f16_t)v.y; h.z = (f16_t)v.z; h.w = (f16_t)v.w;
        *(v4h*)&Xh[i * 4] = h;
    }
}

__global__ __launch_bounds__(256)
void quant_w_f16(const float* __restrict__ W, const float* __restrict__ alphap,
                 f16_t* __restrict__ Wh, long n4)
{
    const float inv_alpha = 1.0f / alphap[0];
    long i = (long)blockIdx.x * 256 + threadIdx.x;
    if (i < n4) {
        v4f v = *(const v4f*)&W[i * 4];
        v4h h;
        h.x = (f16_t)fminf(1.f, fmaxf(-1.f, rintf(v.x * inv_alpha)));
        h.y = (f16_t)fminf(1.f, fmaxf(-1.f, rintf(v.y * inv_alpha)));
        h.z = (f16_t)fminf(1.f, fmaxf(-1.f, rintf(v.z * inv_alpha)));
        h.w = (f16_t)fminf(1.f, fmaxf(-1.f, rintf(v.w * inv_alpha)));
        *(v4h*)&Wh[i * 4] = h;
    }
}

// ---------------------------------------------------------------------------
// Phase 2: f16 WMMA GEMM, 128(M) x 256(N) block tile, K-step 32,
// 8 waves in 2x4 layout, 64x64 per wave, double-buffered LDS fed by
// GLOBAL_LOAD_ASYNC_TO_LDS_B128 (ASYNCcnt pipeline, 1 barrier / iter).
// ---------------------------------------------------------------------------
#define GM   128
#define GN   256
#define GK   32
#define LSTR 40   // padded f16 row stride (80 B): 16B-aligned, bank-conflict free

__global__ __launch_bounds__(256)
void bitnet_gemm_wmma_f16(const f16_t* __restrict__ Xh,
                          const f16_t* __restrict__ Wh,
                          const float* __restrict__ alphap,
                          float* __restrict__ Y,
                          int M, int N, int K)
{
    __shared__ __align__(16) f16_t Xt[2][GM * LSTR];   // 2 x 10 KB
    __shared__ __align__(16) f16_t Wt[2][GN * LSTR];   // 2 x 20 KB

    const int t    = threadIdx.x;
    const int lane = t & 31;
    const int wave = t >> 5;
    const int wm   = wave >> 2;    // 0..1 : 64-row band
    const int wn   = wave & 3;     // 0..3 : 64-col band
    const int lh   = lane >> 4;    // lane half per WMMA operand layout
    const int l16  = lane & 15;

    const long m0 = (long)blockIdx.x * GM;
    const long n0 = (long)blockIdx.y * GN;
    const float alpha = alphap[0];

    // Chunk maps: 16B (8 f16) chunks. X tile: 128 rows x 4 = 512 chunks.
    // W tile: 256 rows x 4 = 1024 chunks. 256 threads.
    const int xr0 = t >> 2;                 // X row 0..63
    const int xr1 = (t >> 2) + 64;          // X row 64..127
    const int xc  = (t & 3) << 3;           // col 0/8/16/24 (f16 units)

    v8f acc[4][4];
#pragma unroll
    for (int i = 0; i < 4; ++i)
#pragma unroll
        for (int j = 0; j < 4; ++j)
            acc[i][j] = v8f{0.f, 0.f, 0.f, 0.f, 0.f, 0.f, 0.f, 0.f};

    // ---- async-stage one K-tile into LDS buffer `buf` ----
    auto stage_tile = [&](int buf, long k) {
        async_load_b128(lds_off(&Xt[buf][xr0 * LSTR + xc]),
                        &Xh[(m0 + xr0) * (long)K + k + xc]);
        async_load_b128(lds_off(&Xt[buf][xr1 * LSTR + xc]),
                        &Xh[(m0 + xr1) * (long)K + k + xc]);
#pragma unroll
        for (int i = 0; i < 4; ++i) {
            const int c   = t + 256 * i;
            const int row = c >> 2;
            const int col = (c & 3) << 3;
            async_load_b128(lds_off(&Wt[buf][row * LSTR + col]),
                            &Wh[(n0 + row) * (long)K + k + col]);
        }
    };

    // ---- prologue: tile 0 ----
    stage_tile(0, 0);
    wait_async0();
    __syncthreads();

    int cur = 0;
    for (int k0 = 0; k0 < K; k0 += GK) {
        const int  nxt       = cur ^ 1;
        const bool have_next = (k0 + GK) < K;

        // issue next-tile async DMA; it lands in LDS while the WMMAs run
        if (have_next) {
            stage_tile(nxt, k0 + GK);
            if (k0 + 2 * GK < K) {   // one 128B line = 64 f16 = two k-steps
                __builtin_prefetch(&Xh[(m0 + (t & 127)) * (long)K + k0 + 2 * GK], 0, 1);
                __builtin_prefetch(&Wh[(n0 + t) * (long)K + k0 + 2 * GK], 0, 1);
            }
        }

        // ---- fragments from current buffer ----
        // A (16x32): lane<16 -> row l16, K {0..7,16..23}; lane>=16 -> K {8..15,24..31}
        v16h af[4];
#pragma unroll
        for (int ti = 0; ti < 4; ++ti) {
            const f16_t* p = &Xt[cur][(wm * 64 + ti * 16 + l16) * LSTR];
            v8h lo = *(const v8h*)&p[lh * 8];
            v8h hi = *(const v8h*)&p[16 + lh * 8];
#pragma unroll
            for (int e = 0; e < 8; ++e) { af[ti][e] = lo[e]; af[ti][e + 8] = hi[e]; }
        }
        // B (32x16): lane<16 -> col l16, K {0..7,16..23}; lane>=16 -> K {8..15,24..31}
        v16h bf[4];
#pragma unroll
        for (int tj = 0; tj < 4; ++tj) {
            const f16_t* p = &Wt[cur][(wn * 64 + tj * 16 + l16) * LSTR];
            v8h lo = *(const v8h*)&p[lh * 8];
            v8h hi = *(const v8h*)&p[16 + lh * 8];
#pragma unroll
            for (int e = 0; e < 8; ++e) { bf[tj][e] = lo[e]; bf[tj][e + 8] = hi[e]; }
        }

        // ---- 16 WMMAs per k-step ----
#pragma unroll
        for (int ti = 0; ti < 4; ++ti)
#pragma unroll
            for (int tj = 0; tj < 4; ++tj)
                acc[ti][tj] = __builtin_amdgcn_wmma_f32_16x16x32_f16(
                    false, af[ti], false, bf[tj],
                    (short)0, acc[ti][tj], false, false);

        // async DMA for tile k+1 must be in LDS before anyone reads it
        wait_async0();
        __syncthreads();
        cur = nxt;
    }

    // ---- epilogue: scale by alpha, fp32 stores coalesced across lanes ----
#pragma unroll
    for (int ti = 0; ti < 4; ++ti)
#pragma unroll
        for (int tj = 0; tj < 4; ++tj) {
            const long mb = m0 + wm * 64 + ti * 16 + lh * 8;
            const long nn = n0 + wn * 64 + tj * 16 + l16;
#pragma unroll
            for (int r = 0; r < 8; ++r)
                Y[(mb + r) * (long)N + nn] = acc[ti][tj][r] * alpha;
        }
}

// ---------------------------------------------------------------------------
// Fallback: fused single-pass kernel when ws too small.
// ---------------------------------------------------------------------------
#define TM   128
#define TN   128
#define TK   32

__global__ __launch_bounds__(256)
void bitnet_gemm_wmma_fused(const float* __restrict__ X,
                            const float* __restrict__ W,
                            const float* __restrict__ alphap,
                            float* __restrict__ Y,
                            int M, int N, int K)
{
    __shared__ __align__(16) f16_t Xt[TM * LSTR];
    __shared__ __align__(16) f16_t Wt[TN * LSTR];

    const int t    = threadIdx.x;
    const int lane = t & 31;
    const int wave = t >> 5;
    const int wm   = wave >> 1;
    const int wn   = wave & 1;
    const int lh   = lane >> 4;
    const int l16  = lane & 15;

    const long m0 = (long)blockIdx.x * TM;
    const long n0 = (long)blockIdx.y * TN;

    const float alpha     = alphap[0];
    const float inv_alpha = 1.0f / alpha;

    v8f acc[2][4];
#pragma unroll
    for (int i = 0; i < 2; ++i)
#pragma unroll
        for (int j = 0; j < 4; ++j)
            acc[i][j] = v8f{0.f, 0.f, 0.f, 0.f, 0.f, 0.f, 0.f, 0.f};

    for (int k0 = 0; k0 < K; k0 += TK) {
#pragma unroll
        for (int i = 0; i < 4; ++i) {
            const int c   = t + 256 * i;
            const int row = c >> 3;
            const int col = (c & 7) << 2;
            v4f v = *(const v4f*)&X[(m0 + row) * (long)K + k0 + col];
            v4h h;
            h.x = (f16_t)v.x; h.y = (f16_t)v.y; h.z = (f16_t)v.z; h.w = (f16_t)v.w;
            *(v4h*)&Xt[row * LSTR + col] = h;
        }
#pragma unroll
        for (int i = 0; i < 4; ++i) {
            const int c   = t + 256 * i;
            const int row = c >> 3;
            const int col = (c & 7) << 2;
            v4f v = *(const v4f*)&W[(n0 + row) * (long)K + k0 + col];
            v4h h;
            h.x = (f16_t)fminf(1.f, fmaxf(-1.f, rintf(v.x * inv_alpha)));
            h.y = (f16_t)fminf(1.f, fmaxf(-1.f, rintf(v.y * inv_alpha)));
            h.z = (f16_t)fminf(1.f, fmaxf(-1.f, rintf(v.z * inv_alpha)));
            h.w = (f16_t)fminf(1.f, fmaxf(-1.f, rintf(v.w * inv_alpha)));
            *(v4h*)&Wt[row * LSTR + col] = h;
        }
        __syncthreads();

        v16h af[2];
#pragma unroll
        for (int ti = 0; ti < 2; ++ti) {
            const f16_t* p = &Xt[(wm * 32 + ti * 16 + l16) * LSTR];
            v8h lo = *(const v8h*)&p[lh * 8];
            v8h hi = *(const v8h*)&p[16 + lh * 8];
#pragma unroll
            for (int e = 0; e < 8; ++e) { af[ti][e] = lo[e]; af[ti][e + 8] = hi[e]; }
        }
        v16h bf[4];
#pragma unroll
        for (int tj = 0; tj < 4; ++tj) {
            const f16_t* p = &Wt[(wn * 64 + tj * 16 + l16) * LSTR];
            v8h lo = *(const v8h*)&p[lh * 8];
            v8h hi = *(const v8h*)&p[16 + lh * 8];
#pragma unroll
            for (int e = 0; e < 8; ++e) { bf[tj][e] = lo[e]; bf[tj][e + 8] = hi[e]; }
        }
#pragma unroll
        for (int ti = 0; ti < 2; ++ti)
#pragma unroll
            for (int tj = 0; tj < 4; ++tj)
                acc[ti][tj] = __builtin_amdgcn_wmma_f32_16x16x32_f16(
                    false, af[ti], false, bf[tj], (short)0, acc[ti][tj], false, false);
        __syncthreads();
    }

#pragma unroll
    for (int ti = 0; ti < 2; ++ti)
#pragma unroll
        for (int tj = 0; tj < 4; ++tj) {
            const long mb = m0 + wm * 32 + ti * 16 + lh * 8;
            const long nn = n0 + wn * 64 + tj * 16 + l16;
#pragma unroll
            for (int r = 0; r < 8; ++r)
                Y[(mb + r) * (long)N + nn] = acc[ti][tj][r] * alpha;
        }
}

// ---------------------------------------------------------------------------
extern "C" void kernel_launch(void* const* d_in, const int* in_sizes, int n_in,
                              void* d_out, int out_size, void* d_ws, size_t ws_size,
                              hipStream_t stream) {
    (void)n_in; (void)out_size;
    const float* X = (const float*)d_in[0];   // [M,K] fp32 (M = B*S)
    const float* W = (const float*)d_in[1];   // [N,K] fp32
    const float* A = (const float*)d_in[2];   // alpha scalar fp32
    float* Y = (float*)d_out;                 // [M,N] fp32

    const int K = 4096;
    const int N = 4096;
    const int M = in_sizes[0] / K;            // 8192 for reference shapes

    const size_t needX = (size_t)M * K * sizeof(f16_t);
    const size_t needW = (size_t)N * K * sizeof(f16_t);

    if (ws_size >= needX + needW) {
        f16_t* Xh = (f16_t*)d_ws;
        f16_t* Wh = (f16_t*)((char*)d_ws + needX);

        const long n4x = (long)M * K / 4;
        const long n4w = (long)N * K / 4;
        cvt_x_f16  <<<(int)(n4x / 256), 256, 0, stream>>>(X, Xh, n4x);
        quant_w_f16<<<(int)(n4w / 256), 256, 0, stream>>>(W, A, Wh, n4w);

        dim3 grid(M / GM, N / GN);            // 64 x 16 blocks
        bitnet_gemm_wmma_f16<<<grid, dim3(256), 0, stream>>>(Xh, Wh, A, Y, M, N, K);
    } else {
        dim3 grid(M / TM, N / TN);            // 64 x 32 blocks
        bitnet_gemm_wmma_fused<<<grid, dim3(256), 0, stream>>>(X, W, A, Y, M, N, K);
    }
}
